// MultiHeadAttention_89369679495714
// MI455X (gfx1250) — matrix operand order repaired and checked by reading, not compiled, not run on hardware
//
#include <hip/hip_runtime.h>
#include <hip/hip_bf16.h>

// ---------------------------------------------------------------------------
// MHA forward for MI455X (gfx1250, wave32, WMMA).
// fp32->bf16 convert | QKV GEMM (async-LDS double-buffered WMMA) |
// flash-attention (64-key chunks, WMMA bf16, f32 online softmax) |
// out-proj GEMM (WMMA bf16, f32 out).
// ---------------------------------------------------------------------------

typedef __bf16 bf16_t;
typedef __attribute__((ext_vector_type(16))) __bf16 v16bf;
typedef __attribute__((ext_vector_type(4)))  __bf16 v4bf;
typedef __attribute__((ext_vector_type(8)))  float  v8f;
typedef __attribute__((ext_vector_type(4)))  int    v4i;

#define D_MODEL 1024
#define NUM_HEADS 16
#define D_HEAD 64
#define BATCH 4
#define SEQ 2048

#if defined(__has_builtin)
#if __has_builtin(__builtin_amdgcn_global_load_async_to_lds_b128)
#define HAVE_ASYNC_LDS 1
#endif
#endif
#ifndef HAVE_ASYNC_LDS
#define HAVE_ASYNC_LDS 0
#endif

union Frag { v16bf v; uint4 u[2]; };

static __device__ __forceinline__ v8f wmma_bf16(v16bf a, v16bf b, v8f c) {
  // D = A(16x32) * B(32x16) + C(16x16 f32)
  return __builtin_amdgcn_wmma_f32_16x16x32_bf16(
      /*neg_a=*/false, a, /*neg_b=*/false, b,
      /*c_mod=*/(short)0, c, /*reuse_a=*/false, /*reuse_b=*/false);
}

// 16-byte global -> LDS copy; async DMA (ASYNCcnt) when the builtin exists.
static __device__ __forceinline__ void async_copy16(bf16_t* dst,
                                                    const bf16_t* src) {
#if HAVE_ASYNC_LDS
  __builtin_amdgcn_global_load_async_to_lds_b128(
      (__attribute__((address_space(1))) v4i*)(src),
      (__attribute__((address_space(3))) v4i*)(dst), 0, 0);
#else
  *(uint4*)dst = *(const uint4*)src;
#endif
}

static __device__ __forceinline__ void wait_async0() {
#if HAVE_ASYNC_LDS
#if __has_builtin(__builtin_amdgcn_s_wait_asynccnt)
  __builtin_amdgcn_s_wait_asynccnt(0);
#else
  asm volatile("s_wait_asynccnt 0" ::: "memory");
#endif
#endif
}

// ---------------------------------------------------------------------------
// fp32 -> bf16 conversion (4 elements per thread)
// ---------------------------------------------------------------------------
__global__ __launch_bounds__(256) void cvt_kernel(const float* __restrict__ in,
                                                  bf16_t* __restrict__ out,
                                                  int n4) {
  int i = blockIdx.x * 256 + threadIdx.x;
  if (i >= n4) return;
  float4 f = ((const float4*)in)[i];
  v4bf o;
  o[0] = (bf16_t)f.x; o[1] = (bf16_t)f.y; o[2] = (bf16_t)f.z; o[3] = (bf16_t)f.w;
  ((v4bf*)out)[i] = o;
}

// ---------------------------------------------------------------------------
// Tiled WMMA GEMM: out[m,n] = sum_k X[m,k]*W[n,k] + bias[n]
// WG: 256 thr (8 waves), tile M=128, N=64, K-step 64; wave tile 32x32.
// Double-buffered LDS tiles filled with async global->LDS DMA.
// MODE 0: QKV epilogue (scatter bf16 q/k/vT, q pre-scaled by 1/sqrt(dh))
// MODE 1: f32 output + bias (final projection)
// ---------------------------------------------------------------------------
template <int MODE>
__global__ __launch_bounds__(256) void gemm_bf16_kernel(
    const bf16_t* __restrict__ X, const bf16_t* __restrict__ W,
    const float* __restrict__ bias, float* __restrict__ outF,
    bf16_t* __restrict__ qbuf, bf16_t* __restrict__ kbuf,
    bf16_t* __restrict__ vtbuf, int Kd, int N) {
  constexpr int LDST = 72;  // 144 B rows = 9x16B -> conflict-free b128
  __shared__ bf16_t Xs[2][128 * LDST];
  __shared__ bf16_t Ws[2][64 * LDST];

  const int lane = threadIdx.x & 31;
  const int wid  = threadIdx.x >> 5;
  const int l16  = lane & 15;
  const int hh   = lane >> 4;
  const int m0   = blockIdx.y * 128;
  const int n0   = blockIdx.x * 64;
  const int moff = (wid >> 1) * 32;  // 4 waves along M
  const int noff = (wid & 1) * 32;   // 2 waves along N

  auto load_tile = [&](int k0, int buf) {
    const int t = threadIdx.x;
#pragma unroll
    for (int i = 0; i < 4; ++i) {  // X: 128 rows x 128 B
      int idx = t + i * 256;
      int row = idx >> 3, ch = idx & 7;
      async_copy16(&Xs[buf][row * LDST + ch * 8],
                   &X[(size_t)(m0 + row) * Kd + k0 + ch * 8]);
    }
#pragma unroll
    for (int i = 0; i < 2; ++i) {  // W: 64 rows x 128 B
      int idx = t + i * 256;
      int row = idx >> 3, ch = idx & 7;
      async_copy16(&Ws[buf][row * LDST + ch * 8],
                   &W[(size_t)(n0 + row) * Kd + k0 + ch * 8]);
    }
  };

  v8f acc[2][2] = {};
  load_tile(0, 0);
  int cur = 0;

  for (int k0 = 0; k0 < Kd; k0 += 64) {
    wait_async0();    // our issued DMA for buf `cur` is done
    __syncthreads();  // everyone's tile visible; prior readers of buf^1 done
    if (k0 + 64 < Kd) load_tile(k0 + 64, cur ^ 1);  // overlap DMA with WMMA

#pragma unroll
    for (int kk = 0; kk < 64; kk += 32) {
      Frag a[2], b[2];
#pragma unroll
      for (int mf = 0; mf < 2; ++mf) {
        // A 16x32 bf16: lane=row(M), elems 0..7 = K 8h..8h+7, 8..15 = 16+8h..
        const bf16_t* p = &Xs[cur][(moff + 16 * mf + l16) * LDST + kk];
        a[mf].u[0] = *(const uint4*)(p + 8 * hh);
        a[mf].u[1] = *(const uint4*)(p + 16 + 8 * hh);
      }
#pragma unroll
      for (int nf = 0; nf < 2; ++nf) {
        // B 32x16 bf16: lane=col(N), elems 0..7 = K 8h..8h+7, 8..15 = 16+8h..
        const bf16_t* p = &Ws[cur][(noff + 16 * nf + l16) * LDST + kk];
        b[nf].u[0] = *(const uint4*)(p + 8 * hh);
        b[nf].u[1] = *(const uint4*)(p + 16 + 8 * hh);
      }
#pragma unroll
      for (int mf = 0; mf < 2; ++mf)
#pragma unroll
        for (int nf = 0; nf < 2; ++nf)
          acc[mf][nf] = wmma_bf16(a[mf].v, b[nf].v, acc[mf][nf]);
    }
    cur ^= 1;
  }

  // Epilogue. C layout: vgpr r -> row r + 8*(lane/16), col = lane%16.
#pragma unroll
  for (int mf = 0; mf < 2; ++mf)
#pragma unroll
    for (int nf = 0; nf < 2; ++nf)
#pragma unroll
      for (int r = 0; r < 8; ++r) {
        int m = m0 + moff + 16 * mf + r + 8 * hh;
        int n = n0 + noff + 16 * nf + l16;
        float val = acc[mf][nf][r] + bias[n];
        if (MODE == 0) {
          int which = n >> 10;          // 0=q 1=k 2=v
          int head  = (n & 1023) >> 6;  // /64
          int dd    = n & 63;
          int bb    = m >> 11;          // /SEQ
          int tt    = m & 2047;
          size_t bh = (size_t)bb * NUM_HEADS + head;
          if (which == 0)
            qbuf[(bh * SEQ + tt) * D_HEAD + dd] = (bf16_t)(val * 0.125f);
          else if (which == 1)
            kbuf[(bh * SEQ + tt) * D_HEAD + dd] = (bf16_t)val;
          else
            vtbuf[(bh * D_HEAD + dd) * SEQ + tt] = (bf16_t)val;
        } else {
          outF[(size_t)m * N + n] = val;
        }
      }
}

// ---------------------------------------------------------------------------
// Flash attention: one wave per (b, h, 16-query tile); stream 64-key chunks
// (16 WMMAs per chunk; softmax shuffle tree amortized over 64 keys).
// Q pre-scaled by 1/sqrt(dh) in the QKV epilogue.
// ---------------------------------------------------------------------------
__global__ __launch_bounds__(256) void attn_kernel(
    const bf16_t* __restrict__ qbuf, const bf16_t* __restrict__ kbuf,
    const bf16_t* __restrict__ vtbuf, bf16_t* __restrict__ ctx) {
  constexpr int PSTR = 68;  // 16x64 f32 P tile, padded: 272 B rows -> no conflicts
  __shared__ float pls[8][16 * PSTR];

  const int lane  = threadIdx.x & 31;
  const int wid   = threadIdx.x >> 5;
  const int l16   = lane & 15;
  const int hh    = lane >> 4;
  const int gw    = blockIdx.x * 8 + wid;  // 0..8191
  const int qtile = gw & (SEQ / 16 - 1);   // 0..127
  const int bh    = gw >> 7;               // 0..63

  const bf16_t* Q  = qbuf  + (size_t)bh * SEQ * D_HEAD;
  const bf16_t* Kp = kbuf  + (size_t)bh * SEQ * D_HEAD;
  const bf16_t* Vp = vtbuf + (size_t)bh * D_HEAD * SEQ;
  float* pl = &pls[wid][0];

  // Q A-fragments: 16 rows x 64 (two K=32 fragments)
  Frag aq[2];
  {
    const bf16_t* qp = Q + (size_t)(qtile * 16 + l16) * D_HEAD;
#pragma unroll
    for (int kk = 0; kk < 2; ++kk) {
      aq[kk].u[0] = *(const uint4*)(qp + 32 * kk + 8 * hh);
      aq[kk].u[1] = *(const uint4*)(qp + 32 * kk + 16 + 8 * hh);
    }
  }

  v8f acc[4] = {};
  float mrow[8], lrow[8];
#pragma unroll
  for (int r = 0; r < 8; ++r) { mrow[r] = -1e30f; lrow[r] = 0.f; }

  for (int j0 = 0; j0 < SEQ; j0 += 64) {
    if (j0 + 64 < SEQ) {  // uniform; per-lane addresses cover next chunk's lines
      __builtin_prefetch(Kp + (size_t)(j0 + 64 + lane) * D_HEAD, 0, 3);
      __builtin_prefetch(Kp + (size_t)(j0 + 96 + lane) * D_HEAD, 0, 3);
      __builtin_prefetch(Vp + (size_t)lane * SEQ + j0 + 64, 0, 3);
      __builtin_prefetch(Vp + (size_t)(32 + lane) * SEQ + j0 + 64, 0, 3);
    }

    // K^T B-fragments: 4 key sub-tiles x 2 dh chunks
    Frag kf[4][2];
#pragma unroll
    for (int nf = 0; nf < 4; ++nf)
#pragma unroll
      for (int kk = 0; kk < 2; ++kk) {
        const bf16_t* kp =
            Kp + (size_t)(j0 + 16 * nf + l16) * D_HEAD + 32 * kk;
        kf[nf][kk].u[0] = *(const uint4*)(kp + 8 * hh);
        kf[nf][kk].u[1] = *(const uint4*)(kp + 16 + 8 * hh);
      }
    v8f s[4];
#pragma unroll
    for (int nf = 0; nf < 4; ++nf) {
      v8f z = {};
      s[nf] = wmma_bf16(aq[0].v, kf[nf][0].v, z);
      s[nf] = wmma_bf16(aq[1].v, kf[nf][1].v, s[nf]);
    }

    // Online softmax; row stats live per (vgpr r, half): row = r + 8*hh.
#pragma unroll
    for (int r = 0; r < 8; ++r) {
      float mx = fmaxf(fmaxf(s[0][r], s[1][r]), fmaxf(s[2][r], s[3][r]));
      mx = fmaxf(mx, __shfl_xor(mx, 1));
      mx = fmaxf(mx, __shfl_xor(mx, 2));
      mx = fmaxf(mx, __shfl_xor(mx, 4));
      mx = fmaxf(mx, __shfl_xor(mx, 8));
      float mnew  = fmaxf(mrow[r], mx);
      float alpha = __expf(mrow[r] - mnew);
      float p0 = __expf(s[0][r] - mnew);
      float p1 = __expf(s[1][r] - mnew);
      float p2 = __expf(s[2][r] - mnew);
      float p3 = __expf(s[3][r] - mnew);
      float ps = (p0 + p1) + (p2 + p3);
      ps += __shfl_xor(ps, 1);
      ps += __shfl_xor(ps, 2);
      ps += __shfl_xor(ps, 4);
      ps += __shfl_xor(ps, 8);
      lrow[r] = lrow[r] * alpha + ps;
      mrow[r] = mnew;
#pragma unroll
      for (int nf = 0; nf < 4; ++nf) acc[nf][r] *= alpha;
      int prow = (r + 8 * hh) * PSTR;
      pl[prow + l16]      = p0;
      pl[prow + 16 + l16] = p1;
      pl[prow + 32 + l16] = p2;
      pl[prow + 48 + l16] = p3;
    }
    asm volatile("s_wait_dscnt 0" ::: "memory");

    // Re-read P (16x64) in A-fragment layout (two K=32 frags), f32->bf16.
    v16bf pa[2];
#pragma unroll
    for (int kk2 = 0; kk2 < 2; ++kk2) {
      const float* pr = pl + l16 * PSTR + 32 * kk2;
      float4 t0 = *(const float4*)(pr + 8 * hh);
      float4 t1 = *(const float4*)(pr + 8 * hh + 4);
      float4 t2 = *(const float4*)(pr + 16 + 8 * hh);
      float4 t3 = *(const float4*)(pr + 16 + 8 * hh + 4);
      pa[kk2][0]  = (bf16_t)t0.x; pa[kk2][1]  = (bf16_t)t0.y;
      pa[kk2][2]  = (bf16_t)t0.z; pa[kk2][3]  = (bf16_t)t0.w;
      pa[kk2][4]  = (bf16_t)t1.x; pa[kk2][5]  = (bf16_t)t1.y;
      pa[kk2][6]  = (bf16_t)t1.z; pa[kk2][7]  = (bf16_t)t1.w;
      pa[kk2][8]  = (bf16_t)t2.x; pa[kk2][9]  = (bf16_t)t2.y;
      pa[kk2][10] = (bf16_t)t2.z; pa[kk2][11] = (bf16_t)t2.w;
      pa[kk2][12] = (bf16_t)t3.x; pa[kk2][13] = (bf16_t)t3.y;
      pa[kk2][14] = (bf16_t)t3.z; pa[kk2][15] = (bf16_t)t3.w;
    }

    // O += P(16x64) * V(64x64): V stored transposed [dh][T] -> K-contiguous B
#pragma unroll
    for (int nf = 0; nf < 4; ++nf) {
      const bf16_t* vp = Vp + (size_t)(16 * nf + l16) * SEQ + j0 + 8 * hh;
      Frag vf0, vf1;
      vf0.u[0] = *(const uint4*)vp;
      vf0.u[1] = *(const uint4*)(vp + 16);
      vf1.u[0] = *(const uint4*)(vp + 32);
      vf1.u[1] = *(const uint4*)(vp + 48);
      acc[nf] = wmma_bf16(pa[0], vf0.v, acc[nf]);
      acc[nf] = wmma_bf16(pa[1], vf1.v, acc[nf]);
    }
  }

  // Normalize and store ctx[b, t, h*64 + d] (bf16)
  const int bb   = bh >> 4;
  const int head = bh & 15;
  float linv[8];
#pragma unroll
  for (int r = 0; r < 8; ++r) linv[r] = 1.f / lrow[r];
#pragma unroll
  for (int nf = 0; nf < 4; ++nf)
#pragma unroll
    for (int r = 0; r < 8; ++r) {
      int q = qtile * 16 + r + 8 * hh;
      int d = 16 * nf + l16;
      float val = acc[nf][r] * linv[r];
      ctx[((size_t)bb * SEQ + q) * D_MODEL + head * D_HEAD + d] = (bf16_t)val;
    }
}

// ---------------------------------------------------------------------------
extern "C" void kernel_launch(void* const* d_in, const int* in_sizes, int n_in,
                              void* d_out, int out_size, void* d_ws,
                              size_t ws_size, hipStream_t stream) {
  (void)in_sizes; (void)n_in; (void)out_size; (void)ws_size;
  const float* x     = (const float*)d_in[0];
  const float* qkv_w = (const float*)d_in[1];
  const float* qkv_b = (const float*)d_in[2];
  const float* out_w = (const float*)d_in[3];
  const float* out_b = (const float*)d_in[4];
  float* out = (float*)d_out;

  const size_t XE = (size_t)BATCH * SEQ * D_MODEL;
  const size_t WQ = (size_t)3 * D_MODEL * D_MODEL;
  const size_t WO = (size_t)D_MODEL * D_MODEL;
  const size_t HE = (size_t)BATCH * NUM_HEADS * SEQ * D_HEAD;
  const size_t CT = XE;

  char* p = (char*)d_ws;
  bf16_t* xb  = (bf16_t*)p; p += XE * 2;
  bf16_t* wqb = (bf16_t*)p; p += WQ * 2;
  bf16_t* wob = (bf16_t*)p; p += WO * 2;
  bf16_t* qb  = (bf16_t*)p; p += HE * 2;
  bf16_t* kb  = (bf16_t*)p; p += HE * 2;
  bf16_t* vtb = (bf16_t*)p; p += HE * 2;
  bf16_t* ctx = (bf16_t*)p; p += CT * 2;

  cvt_kernel<<<(int)(XE / 4 / 256), 256, 0, stream>>>(x, xb, (int)(XE / 4));
  cvt_kernel<<<(int)(WQ / 4 / 256), 256, 0, stream>>>(qkv_w, wqb, (int)(WQ / 4));
  cvt_kernel<<<(int)(WO / 4 / 256), 256, 0, stream>>>(out_w, wob, (int)(WO / 4));

  dim3 g1(3 * D_MODEL / 64, BATCH * SEQ / 128);  // (48, 64)
  gemm_bf16_kernel<0><<<g1, 256, 0, stream>>>(xb, wqb, qkv_b, nullptr, qb, kb,
                                              vtb, D_MODEL, 3 * D_MODEL);

  attn_kernel<<<BATCH * NUM_HEADS * (SEQ / 16) / 8, 256, 0, stream>>>(qb, kb,
                                                                      vtb, ctx);

  dim3 g2(D_MODEL / 64, BATCH * SEQ / 128);  // (16, 64)
  gemm_bf16_kernel<1><<<g2, 256, 0, stream>>>(ctx, wob, out_b, out, nullptr,
                                              nullptr, nullptr, D_MODEL,
                                              D_MODEL);
}
